// TriangleMultiplication_33913061769435
// MI455X (gfx1250) — compile-verified, matrix operand order checked by reading, not run on hardware
//
#include <hip/hip_runtime.h>

typedef __attribute__((ext_vector_type(16))) __bf16 v16bf;
typedef __attribute__((ext_vector_type(8)))  float  v8f;

#define NN 262144   // N*N
#define NSZ 512
#define CC 128
#define EPSF 1e-5f

// ---------- helpers ----------

__device__ inline v8f zero8() {
    v8f z = {0.f, 0.f, 0.f, 0.f, 0.f, 0.f, 0.f, 0.f};
    return z;
}

__device__ inline v8f wmma_bf16(v16bf a, v16bf b, v8f c) {
    // D = A(16x32 bf16) * B(32x16 bf16) + C(16x16 f32)
    return __builtin_amdgcn_wmma_f32_16x16x32_bf16(
        /*neg_a=*/false, a, /*neg_b=*/false, b,
        /*c_mod=*/(short)0, c, /*reuse_a=*/false, /*reuse_b=*/false);
}

// Load a 16-element bf16 fragment for one lane.
// Per ISA layout (16-bit A/B operand): this lane supplies K = kb..kb+7 and
// kb+16..kb+23 of its row -> two contiguous 16B loads.
__device__ inline v16bf load_bfrag(const __bf16* rowptr, int kb) {
    v16bf f;
    reinterpret_cast<uint4*>(&f)[0] = *reinterpret_cast<const uint4*>(rowptr + kb);
    reinterpret_cast<uint4*>(&f)[1] = *reinterpret_cast<const uint4*>(rowptr + kb + 16);
    return f;
}

// Row layernorm stats: lane pair (r, r+16) each sums 64 of the 128 elements,
// combined with a cross-half shuffle (wave32).
__device__ inline void row_stats(const float* rowptr, int lane, float& mean, float& rstd) {
    int half = lane >> 4;
    const float* p = rowptr + half * 64;
    float s = 0.f, sq = 0.f;
#pragma unroll
    for (int t = 0; t < 16; ++t) {
        float4 v = *reinterpret_cast<const float4*>(p + t * 4);
        s  += v.x + v.y + v.z + v.w;
        sq += v.x * v.x + v.y * v.y + v.z * v.z + v.w * v.w;
    }
    s  += __shfl_xor(s, 16, 32);
    sq += __shfl_xor(sq, 16, 32);
    mean = s * (1.f / 128.f);
    float var = sq * (1.f / 128.f) - mean * mean;
    rstd = rsqrtf(var + EPSF);
}

// Build a bf16 A-fragment from a layernormed fp32 row.
__device__ inline v16bf make_ln_frag(const float* rowptr, int kb, float mean, float rstd,
                                     const float* __restrict__ scale,
                                     const float* __restrict__ bias) {
    v16bf f;
#pragma unroll
    for (int g = 0; g < 2; ++g) {
        int k0 = kb + g * 16;
        float4 v0 = *reinterpret_cast<const float4*>(rowptr + k0);
        float4 v1 = *reinterpret_cast<const float4*>(rowptr + k0 + 4);
        float vv[8] = {v0.x, v0.y, v0.z, v0.w, v1.x, v1.y, v1.z, v1.w};
#pragma unroll
        for (int e = 0; e < 8; ++e) {
            int k = k0 + e;
            float xv = (vv[e] - mean) * rstd * scale[k] + bias[k];
            f[g * 8 + e] = (__bf16)xv;
        }
    }
    return f;
}

__device__ inline float sigmoidf_(float x) { return 1.0f / (1.0f + __expf(-x)); }

// ---------- kernel 0: weight transpose + fp32 -> bf16 ----------
// wt[d*K + k] = (bf16) w[k*D + d]   (so B-operand fragment loads are contiguous)
__global__ void k0_wt(const float* __restrict__ w, __bf16* __restrict__ wt, int K, int D) {
    int idx = blockIdx.x * blockDim.x + threadIdx.x;
    if (idx >= K * D) return;
    int d = idx / K;
    int k = idx - d * K;
    wt[idx] = (__bf16)w[(size_t)k * D + d];
}

// ---------- kernel 1: LN1 + dual projection GEMM + mask*sigmoid gating ----------
// Per wave: 16 rows of x (rows of the flattened (i,j) index), K=128.
// Output: a_ws[c][row], b_ws[c][row] as bf16 (channel-major for the NT GEMM).
__global__ void __launch_bounds__(256)
k1_ln_projgate(const float* __restrict__ act, const float* __restrict__ mask,
               const float* __restrict__ s1, const float* __restrict__ b1,
               const __bf16* __restrict__ wtp, const __bf16* __restrict__ wtg,
               __bf16* __restrict__ a_ws, __bf16* __restrict__ b_ws) {
    int wave = threadIdx.x >> 5, lane = threadIdx.x & 31;
    int r0 = (blockIdx.x * 8 + wave) * 16;
    int row = lane & 15, half = lane >> 4;

    const float* rowp = act + (size_t)(r0 + row) * CC;
    float mean, rstd;
    row_stats(rowp, lane, mean, rstd);

    v16bf xf[4];
#pragma unroll
    for (int ks = 0; ks < 4; ++ks)
        xf[ks] = make_ln_frag(rowp, ks * 32 + half * 8, mean, rstd, s1, b1);

    float mv[8];
#pragma unroll
    for (int vr = 0; vr < 8; ++vr)
        mv[vr] = mask[r0 + half * 8 + vr];   // D-layout: M = vr + 8*half

#pragma unroll 2
    for (int nt = 0; nt < 16; ++nt) {        // 256 output channels, 16 at a time
        int d0 = nt * 16;
        const __bf16* bprow = wtp + (size_t)(d0 + row) * CC;
        const __bf16* bgrow = wtg + (size_t)(d0 + row) * CC;
        v8f ap = zero8(), ag = zero8();
#pragma unroll
        for (int ks = 0; ks < 4; ++ks) {
            int kb = ks * 32 + half * 8;
            ap = wmma_bf16(xf[ks], load_bfrag(bprow, kb), ap);
            ag = wmma_bf16(xf[ks], load_bfrag(bgrow, kb), ag);
        }
        int d = d0 + row;                    // this lane's output channel
        int kc = d >> 1;                     // a = even d, b = odd d
        __bf16* dst = ((d & 1) ? b_ws : a_ws) + (size_t)kc * NN + r0 + 8 * half;
#pragma unroll
        for (int vr = 0; vr < 8; ++vr) {
            float val = ap[vr] * mv[vr] * sigmoidf_(ag[vr]);
            dst[vr] = (__bf16)val;
        }
    }
}

// ---------- kernel 2: batched NT GEMM  out[c] = a[c] * b[c]^T ----------
// Block tile 128x128 (8 waves in a 4x2 grid); wave tile 32x64 (8 accumulators):
// per K-step 12 b128 loads feed 8 WMMAs; K-loop unrolled 4x so the clause
// scheduler can run loads several WMMAs ahead. grid.x = 128 ch * 16 = 2048.
__global__ void __launch_bounds__(256)
k2_tri(const __bf16* __restrict__ a_ws, const __bf16* __restrict__ b_ws,
       float* __restrict__ out_t) {
    int wave = threadIdx.x >> 5, lane = threadIdx.x & 31;
    int c = blockIdx.x >> 4;
    int rem = blockIdx.x & 15;
    int mt = rem >> 2, nt = rem & 3;
    int m0 = mt * 128 + (wave >> 1) * 32;    // wave grid: 4 (M) x 2 (N)
    int n0 = nt * 128 + (wave & 1) * 64;
    int row = lane & 15, half = lane >> 4;

    const __bf16* abase = a_ws + (size_t)c * NN;
    const __bf16* bbase = b_ws + (size_t)c * NN;
    const __bf16* arow0 = abase + (size_t)(m0 + row) * NSZ;
    const __bf16* arow1 = abase + (size_t)(m0 + 16 + row) * NSZ;
    const __bf16* brow[4];
#pragma unroll
    for (int ns = 0; ns < 4; ++ns)
        brow[ns] = bbase + (size_t)(n0 + ns * 16 + row) * NSZ;

    v8f acc0[4], acc1[4];
#pragma unroll
    for (int ns = 0; ns < 4; ++ns) { acc0[ns] = zero8(); acc1[ns] = zero8(); }

#pragma unroll 4
    for (int ks = 0; ks < 16; ++ks) {        // K = 512 = 16 * 32
        int kb = ks * 32 + half * 8;
        v16bf a0 = load_bfrag(arow0, kb);
        v16bf a1 = load_bfrag(arow1, kb);
#pragma unroll
        for (int ns = 0; ns < 4; ++ns) {
            v16bf bf = load_bfrag(brow[ns], kb);
            acc0[ns] = wmma_bf16(a0, bf, acc0[ns]);
            acc1[ns] = wmma_bf16(a1, bf, acc1[ns]);
        }
    }

#pragma unroll
    for (int ns = 0; ns < 4; ++ns) {
        int ng = n0 + ns * 16 + row;
        int mgb0 = m0 + 8 * half;
        int mgb1 = m0 + 16 + 8 * half;
#pragma unroll
        for (int vr = 0; vr < 8; ++vr) {
            out_t[((size_t)(mgb0 + vr) * NSZ + ng) * CC + c] = acc0[ns][vr];
            out_t[((size_t)(mgb1 + vr) * NSZ + ng) * CC + c] = acc1[ns][vr];
        }
    }
}

// ---------- kernel 3: LN2 + @w_out, gated by sigmoid(LN1(act) @ w_gating) ----------
__global__ void __launch_bounds__(256)
k3_out(const float* __restrict__ out_t, const float* __restrict__ act,
       const float* __restrict__ s2, const float* __restrict__ b2,
       const float* __restrict__ s1, const float* __restrict__ b1,
       const __bf16* __restrict__ wto, const __bf16* __restrict__ wtga,
       float* __restrict__ out) {
    int wave = threadIdx.x >> 5, lane = threadIdx.x & 31;
    int r0 = (blockIdx.x * 8 + wave) * 16;
    int row = lane & 15, half = lane >> 4;

    const float* yrow = out_t + (size_t)(r0 + row) * CC;
    const float* xrow = act + (size_t)(r0 + row) * CC;
    float ym, yr, xm, xr;
    row_stats(yrow, lane, ym, yr);
    row_stats(xrow, lane, xm, xr);

    v16bf yf[4], xf[4];
#pragma unroll
    for (int ks = 0; ks < 4; ++ks) {
        int kb = ks * 32 + half * 8;
        yf[ks] = make_ln_frag(yrow, kb, ym, yr, s2, b2);
        xf[ks] = make_ln_frag(xrow, kb, xm, xr, s1, b1);
    }

#pragma unroll 2
    for (int nt = 0; nt < 8; ++nt) {         // 128 output channels, 16 at a time
        int d0 = nt * 16;
        const __bf16* orow = wto  + (size_t)(d0 + row) * CC;
        const __bf16* grow = wtga + (size_t)(d0 + row) * CC;
        v8f ao = zero8(), ag = zero8();
#pragma unroll
        for (int ks = 0; ks < 4; ++ks) {
            int kb = ks * 32 + half * 8;
            ao = wmma_bf16(yf[ks], load_bfrag(orow, kb), ao);
            ag = wmma_bf16(xf[ks], load_bfrag(grow, kb), ag);
        }
#pragma unroll
        for (int vr = 0; vr < 8; ++vr) {
            int m = 8 * half + vr;
            out[(size_t)(r0 + m) * CC + d0 + row] = ao[vr] * sigmoidf_(ag[vr]);
        }
    }
}

// ---------- launch ----------
extern "C" void kernel_launch(void* const* d_in, const int* in_sizes, int n_in,
                              void* d_out, int out_size, void* d_ws, size_t ws_size,
                              hipStream_t stream) {
    const float* act      = (const float*)d_in[0];
    const float* mask     = (const float*)d_in[1];
    const float* ln1_s    = (const float*)d_in[2];
    const float* ln1_b    = (const float*)d_in[3];
    const float* w_proj   = (const float*)d_in[4];
    const float* w_gate   = (const float*)d_in[5];
    const float* ln2_s    = (const float*)d_in[6];
    const float* ln2_b    = (const float*)d_in[7];
    const float* w_out    = (const float*)d_in[8];
    const float* w_gating = (const float*)d_in[9];
    float* out = (float*)d_out;

    char* ws = (char*)d_ws;
    __bf16* wtp  = (__bf16*)(ws);                       // 256x128 bf16 = 64 KB
    __bf16* wtg  = (__bf16*)(ws + (64 << 10));          // 64 KB
    __bf16* wto  = (__bf16*)(ws + (128 << 10));         // 128x128 bf16 = 32 KB
    __bf16* wtga = (__bf16*)(ws + (160 << 10));         // 32 KB
    __bf16* a_ws = (__bf16*)(ws + (192 << 10));         // 128*512*512 bf16 = 64 MB
    __bf16* b_ws = (__bf16*)(ws + (192 << 10) + ((size_t)64 << 20));
    float*  outt = (float*)(ws + (192 << 10) + ((size_t)128 << 20)); // 128 MB fp32

    // weight transposes (tiny)
    k0_wt<<<(128 * 256 + 255) / 256, 256, 0, stream>>>(w_proj,   wtp,  CC, 2 * CC);
    k0_wt<<<(128 * 256 + 255) / 256, 256, 0, stream>>>(w_gate,   wtg,  CC, 2 * CC);
    k0_wt<<<(128 * 128 + 255) / 256, 256, 0, stream>>>(w_out,    wto,  CC, CC);
    k0_wt<<<(128 * 128 + 255) / 256, 256, 0, stream>>>(w_gating, wtga, CC, CC);

    // stage 1: LN1 + proj/gate + gating -> a,b (bf16, channel-major)
    k1_ln_projgate<<<NN / (8 * 16), 256, 0, stream>>>(act, mask, ln1_s, ln1_b,
                                                      wtp, wtg, a_ws, b_ws);
    // stage 2: 128 batched 512x512x512 NT GEMMs
    k2_tri<<<128 * 4 * 4, 256, 0, stream>>>(a_ws, b_ws, outt);
    // stage 3: LN2 + w_out + sigmoid(x @ w_gating)
    k3_out<<<NN / (8 * 16), 256, 0, stream>>>(outt, act, ln2_s, ln2_b, ln1_s, ln1_b,
                                              wto, wtga, out);
}
